// WaveNet_72172630442541
// MI455X (gfx1250) — compile-verified, matrix operand order checked by reading, not run on hardware
//
#include <hip/hip_runtime.h>
#include <hip/hip_bf16.h>

// ---------------------------------------------------------------------------
// WaveNet teacher-forced forward, CDNA5 (gfx1250, wave32) WMMA implementation.
//   - cond kept at 400-frame resolution; per-block 3x128 LDS stage (t/75)
//   - all matmuls on v_wmma_f32_16x16x32_f16, f32 accumulate
//   - 4 independent WMMA accumulator chains per wave (hides D->C hazards)
//   - branch-free gate (v_exp/v_rcp), global_prefetch warms skip accumulator
// ---------------------------------------------------------------------------

typedef __attribute__((ext_vector_type(16))) _Float16 v16h;
typedef __attribute__((ext_vector_type(8)))  _Float16 v8h;
typedef __attribute__((ext_vector_type(8)))  float    v8f;

namespace {
constexpr int kB      = 2;
constexpr int kT      = 30000;
constexpr int kTpad   = 30208;   // multiple of 128
constexpr int kC4     = 404;     // padded 400-frame cond axis (t/75 <= 402)
constexpr int kLayers = 24;
constexpr int XR      = 72;      // LDS row stride (f16) for 64-wide rows
constexpr int SR      = 264;     // LDS row stride (f16) for 256-wide rows
}

union F16Frag { v16h v; v8h h[2]; };

// Fragment load matching CDNA5 16-bit A/B layouts:
// lane = (mn | hi<<4); element j<8 -> k = j + 8*hi ; j>=8 -> k = 16 + (j-8) + 8*hi
__device__ __forceinline__ v16h load_frag(const _Float16* base, int stride, int lane) {
  const int mn = lane & 15, hi = lane >> 4;
  const _Float16* p = base + mn * stride + 8 * hi;
  F16Frag f;
  f.h[0] = *(const v8h*)(p);
  f.h[1] = *(const v8h*)(p + 16);
  return f.v;
}

__device__ __forceinline__ v8f wmma_f16(v16h a, v16h b, v8f c) {
  return __builtin_amdgcn_wmma_f32_16x16x32_f16(false, a, false, b, (short)0, c, false, false);
}

// branch-free gate: tanh(a)*sigmoid(b) via v_exp_f32 / v_rcp_f32
__device__ __forceinline__ float gate_act(float a, float b) {
  float ac = fminf(fmaxf(a, -15.f), 15.f);
  float ea = __expf(-2.f * ac);
  float th = (1.f - ea) * __builtin_amdgcn_rcpf(1.f + ea);
  float sg = __builtin_amdgcn_rcpf(1.f + __expf(-b));
  return th * sg;
}

// ---------------------------------------------------------------------------
// small elementwise / prep kernels
// ---------------------------------------------------------------------------
__global__ void k_cvt(const float* __restrict__ s, _Float16* __restrict__ d, int n) {
  int i = blockIdx.x * 256 + threadIdx.x;
  if (i < n) d[i] = (_Float16)s[i];
}

// dilate_w [24][128][64][2] -> [24][2][128][64] f16
__global__ void k_cvt_dil(const float* __restrict__ s, _Float16* __restrict__ d, int n) {
  int i = blockIdx.x * 256 + threadIdx.x;
  if (i >= n) return;
  int tap = i & 1, c = (i >> 1) & 63, o = (i >> 7) & 127, L = i >> 14;
  d[(((L * 2 + tap) * 128) + o) * 64 + c] = (_Float16)s[i];
}

__global__ void k_sumsb(const float* __restrict__ skip_b, float* __restrict__ sumsb) {
  int c = threadIdx.x;                     // 256 threads
  float s = 0.f;
  for (int i = 0; i < kLayers; ++i) s += skip_b[i * 256 + c];
  sumsb[c] = s;
}

// up[b][o][t4] = up_b[o] + sum_i lf[b][l][i] * W[i][o][k],  t4 = 4l+k
__global__ void k_up(const float* __restrict__ lf, const float* __restrict__ w,
                     const float* __restrict__ bias, float* __restrict__ up) {
  int i = blockIdx.x * 256 + threadIdx.x;
  if (i >= kB * 80 * 400) return;
  int t4 = i % 400, o = (i / 400) % 80, b = i / (400 * 80);
  int l = t4 >> 2, k = t4 & 3;
  float s = bias[o];
  for (int c = 0; c < 80; ++c)
    s += lf[(b * 100 + l) * 80 + c] * w[(c * 80 + o) * 4 + k];
  up[(b * 80 + o) * 400 + t4] = s;
}

// cond[b][oc][t4] = proj_b[oc] + sum_i proj_w[oc][i] * up[b][i][t4]  (0 in pad)
__global__ void k_cond(const float* __restrict__ pw, const float* __restrict__ pb,
                       const float* __restrict__ up, float* __restrict__ cond) {
  int i = blockIdx.x * 256 + threadIdx.x;
  if (i >= kB * 3072 * kC4) return;
  int t4 = i % kC4, oc = (i / kC4) % 3072, b = i / (kC4 * 3072);
  float s = 0.f;
  if (t4 < 400) {
    s = pb[oc];
    for (int c = 0; c < 80; ++c) s += pw[oc * 80 + c] * up[(b * 80 + c) * 400 + t4];
  }
  cond[i] = s;
}

// x0[b][t][c] = embed[gold[b][t]][c] (f16), zero in pad
__global__ void k_embed(const float* __restrict__ emb, const int* __restrict__ gold,
                        _Float16* __restrict__ x) {
  int i = blockIdx.x * 256 + threadIdx.x;
  if (i >= kB * kTpad * 64) return;
  int c = i & 63, t = (i >> 6) % kTpad, b = i / (kTpad * 64);
  float v = 0.f;
  if (t < kT) v = emb[gold[b * kT + t] * 64 + c];
  x[i] = (_Float16)v;
}

__global__ void k_init_skip(const float* __restrict__ sumsb, float* __restrict__ cum) {
  int i = blockIdx.x * 256 + threadIdx.x;
  if (i >= kB * 256 * kTpad) return;
  cum[i] = sumsb[(i / kTpad) % 256];
}

// ---------------------------------------------------------------------------
// fused per-layer kernel: dilated conv + gate + skip accumulate + residual
// grid: (kTpad/128, B), block 256 (8 waves)
// ---------------------------------------------------------------------------
__global__ __launch_bounds__(256)
void k_layer(const _Float16* __restrict__ xin, _Float16* __restrict__ xout,
             const float* __restrict__ cond, float* __restrict__ cum_skip,
             const _Float16* __restrict__ wdil,   // [2][128][64]
             const float* __restrict__ dil_b,     // [128]
             const _Float16* __restrict__ wskip,  // [256][64]
             const _Float16* __restrict__ wres,   // [64][64]
             const float* __restrict__ res_b,     // [64]
             int dil, int has_res, int layer) {
  __shared__ _Float16 xc[128][XR];   // current tile [t][ch]
  __shared__ _Float16 xp[128][XR];   // t-dil tile; reused as x_new after gate
  __shared__ _Float16 zl[128][XR];   // gated activation
  __shared__ float cond_s[3][128];   // cond(frame,ch) + dil_b  (<=3 frames/block)
  __shared__ float rb_s[64];
  const int tid = threadIdx.x, lane = tid & 31, wave = tid >> 5;
  const int n_ = lane & 15, hi = lane >> 4;
  const int b = blockIdx.y, t0 = blockIdx.x * 128;
  const int c4base = t0 / 75;
  const int brk1 = (c4base + 1) * 75, brk2 = (c4base + 2) * 75;
  const float* cl = cond + ((size_t)b * 3072 + (size_t)layer * 128) * kC4;

  { // warm L0/L2 for the cum_skip tile we will read-modify-write (global_prefetch_b8)
    const float* p = cum_skip + ((size_t)b * 256 + tid) * kTpad + t0;
    __builtin_prefetch(p, 0, 3);
    __builtin_prefetch(p + 32, 0, 3);
    __builtin_prefetch(p + 64, 0, 3);
    __builtin_prefetch(p + 96, 0, 3);
  }

  { // cooperative tile load: 2 threads per row, 32 f16 each (coalesced 128B rows)
    int r = tid >> 1, half = tid & 1;
    const _Float16* src = xin + ((size_t)b * kTpad + (t0 + r)) * 64 + half * 32;
    _Float16* dst = &xc[r][half * 32];
    for (int q = 0; q < 4; ++q) ((v8h*)dst)[q] = ((const v8h*)src)[q];
    int tp = t0 + r - dil;
    _Float16* dst2 = &xp[r][half * 32];
    if (tp >= 0) {
      const _Float16* s2 = xin + ((size_t)b * kTpad + tp) * 64 + half * 32;
      for (int q = 0; q < 4; ++q) ((v8h*)dst2)[q] = ((const v8h*)s2)[q];
    } else {
      v8h z = {};
      for (int q = 0; q < 4; ++q) ((v8h*)dst2)[q] = z;
    }
  }
  // stage cond(+bias) frames and res bias into LDS
  for (int i2 = tid; i2 < 384; i2 += 256) {
    int j = i2 >> 7, m = i2 & 127;
    int c4 = c4base + j; if (c4 > kC4 - 1) c4 = kC4 - 1;
    cond_s[j][m] = cl[(size_t)m * kC4 + c4] + dil_b[m];
  }
  if (tid < 64) rb_s[tid] = has_res ? res_b[tid] : 0.f;
  __syncthreads();

  // ---- y = Wp@x[t-d] + Wc@x[t] + b + cond ; z = tanh(y_lo)*sigmoid(y_hi) ----
  // wave -> m-pair (pr, pr+4) so tanh/sigmoid halves pair up lane/element-wise.
  // Two N-tiles in flight -> 4 independent WMMA accumulator chains.
  {
    const int pr = wave & 3, nh = wave >> 2;
    v16h Ad[4][2];  // [s = tap*2+ks][q]
    for (int tap = 0; tap < 2; ++tap)
      for (int ks = 0; ks < 2; ++ks)
        for (int q = 0; q < 2; ++q)
          Ad[tap * 2 + ks][q] =
              load_frag(wdil + ((size_t)(tap * 128 + (pr + q * 4) * 16)) * 64 + ks * 32, 64, lane);
    for (int ntp = 0; ntp < 2; ++ntp) {
      const int nt0 = nh * 4 + ntp * 2;
      v16h Bf[2][4];  // [u][s]: Bp0,Bp1,Bc0,Bc1
      for (int u = 0; u < 2; ++u) {
        int nt = nt0 + u;
        Bf[u][0] = load_frag(&xp[nt * 16][0],  XR, lane);
        Bf[u][1] = load_frag(&xp[nt * 16][32], XR, lane);
        Bf[u][2] = load_frag(&xc[nt * 16][0],  XR, lane);
        Bf[u][3] = load_frag(&xc[nt * 16][32], XR, lane);
      }
      v8f C[2][2] = {};
#pragma unroll
      for (int s = 0; s < 4; ++s)
#pragma unroll
        for (int u = 0; u < 2; ++u)
#pragma unroll
          for (int q = 0; q < 2; ++q)
            C[u][q] = wmma_f16(Ad[s][q], Bf[u][s], C[u][q]);
      for (int u = 0; u < 2; ++u) {
        int nt = nt0 + u;
        int tt = t0 + nt * 16 + n_;
        int rel = (tt >= brk1 ? 1 : 0) + (tt >= brk2 ? 1 : 0);
#pragma unroll
        for (int r = 0; r < 8; ++r) {
          int mo = r + 8 * hi;
          float ylo = C[u][0][r] + cond_s[rel][pr * 16 + mo];
          float yhi = C[u][1][r] + cond_s[rel][64 + pr * 16 + mo];
          zl[nt * 16 + n_][pr * 16 + mo] = (_Float16)gate_act(ylo, yhi);
        }
      }
    }
  }
  __syncthreads();

  // ---- cum_skip += skip_w @ z  (C loaded/stored in global f32; 4 chains) ----
  {
    v16h A0[2], A1[2];
    for (int q = 0; q < 2; ++q) {
      int mt = wave * 2 + q;
      A0[q] = load_frag(wskip + (size_t)mt * 16 * 64 + 0,  64, lane);
      A1[q] = load_frag(wskip + (size_t)mt * 16 * 64 + 32, 64, lane);
    }
    for (int ntp = 0; ntp < 4; ++ntp) {
      v16h B0[2], B1[2];
      for (int u = 0; u < 2; ++u) {
        int nt = ntp * 2 + u;
        B0[u] = load_frag(&zl[nt * 16][0],  XR, lane);
        B1[u] = load_frag(&zl[nt * 16][32], XR, lane);
      }
      float* cs[2][2];
      v8f c[2][2];
      for (int u = 0; u < 2; ++u)
        for (int q = 0; q < 2; ++q) {
          int mt = wave * 2 + q, nt = ntp * 2 + u;
          cs[u][q] = cum_skip + ((size_t)b * 256 + mt * 16) * kTpad + t0 + nt * 16;
#pragma unroll
          for (int r = 0; r < 8; ++r) c[u][q][r] = cs[u][q][(size_t)(r + 8 * hi) * kTpad + n_];
        }
#pragma unroll
      for (int u = 0; u < 2; ++u)
#pragma unroll
        for (int q = 0; q < 2; ++q)
          c[u][q] = wmma_f16(A0[q], B0[u], c[u][q]);
#pragma unroll
      for (int u = 0; u < 2; ++u)
#pragma unroll
        for (int q = 0; q < 2; ++q)
          c[u][q] = wmma_f16(A1[q], B1[u], c[u][q]);
      for (int u = 0; u < 2; ++u)
        for (int q = 0; q < 2; ++q)
#pragma unroll
          for (int r = 0; r < 8; ++r)
            cs[u][q][(size_t)(r + 8 * hi) * kTpad + n_] = c[u][q][r];
    }
  }

  // ---- x_new = x + res_w @ z + res_b (except last layer; 2 chains) ----
  if (has_res) {
    const int mt = wave & 3, nbase = (wave >> 2) * 4;
    v16h A0 = load_frag(wres + (size_t)mt * 16 * 64 + 0,  64, lane);
    v16h A1 = load_frag(wres + (size_t)mt * 16 * 64 + 32, 64, lane);
    for (int ntp = 0; ntp < 2; ++ntp) {
      v16h B0[2], B1[2];
      v8f c[2];
      for (int u = 0; u < 2; ++u) {
        int nt = nbase + ntp * 2 + u;
        B0[u] = load_frag(&zl[nt * 16][0],  XR, lane);
        B1[u] = load_frag(&zl[nt * 16][32], XR, lane);
#pragma unroll
        for (int r = 0; r < 8; ++r) {
          int m = mt * 16 + r + 8 * hi;
          c[u][r] = (float)xc[nt * 16 + n_][m] + rb_s[m];
        }
      }
#pragma unroll
      for (int u = 0; u < 2; ++u) c[u] = wmma_f16(A0, B0[u], c[u]);
#pragma unroll
      for (int u = 0; u < 2; ++u) c[u] = wmma_f16(A1, B1[u], c[u]);
      for (int u = 0; u < 2; ++u) {
        int nt = nbase + ntp * 2 + u;
#pragma unroll
        for (int r = 0; r < 8; ++r)
          xp[nt * 16 + n_][mt * 16 + r + 8 * hi] = (_Float16)c[u][r];  // xp reused as x_new
      }
    }
  }
  __syncthreads();
  if (has_res) { // coalesced write-out of x_new
    int r = tid >> 1, half = tid & 1;
    _Float16* dst = xout + ((size_t)b * kTpad + (t0 + r)) * 64 + half * 32;
    const _Float16* src = &xp[r][half * 32];
    for (int q = 0; q < 4; ++q) ((v8h*)dst)[q] = ((const v8h*)src)[q];
  }
}

// ---------------------------------------------------------------------------
// output head: h1 = w1 @ relu(skip); h2 = w2 @ relu(h1); log_softmax(ch)
// grid: (kTpad/64, B), block 256 (8 waves); A hoisted over 4 N-tile chains
// ---------------------------------------------------------------------------
__global__ __launch_bounds__(256)
void k_out(const float* __restrict__ cum_skip, const _Float16* __restrict__ w1,
           const _Float16* __restrict__ w2, float* __restrict__ out) {
  __shared__ _Float16 s1[64][SR];
  __shared__ _Float16 s2[64][SR];
  __shared__ float    h2[64][SR];
  const int tid = threadIdx.x, lane = tid & 31, wave = tid >> 5;
  const int n_ = lane & 15, hi = lane >> 4;
  const int b = blockIdx.y, t0 = blockIdx.x * 64;

  { // stage 0: transpose relu(cum_skip) tile -> LDS f16 [t][ch]
    const float* csp = cum_skip + ((size_t)b * 256 + tid) * kTpad + t0;
    for (int t = 0; t < 64; ++t) s1[t][tid] = (_Float16)fmaxf(csp[t], 0.f);
  }
  __syncthreads();

  // stage 1: h1 = w1 @ s1 (M=256, N=64, K=256); 4 accumulator chains per wave
  for (int q = 0; q < 2; ++q) {
    int mt = wave * 2 + q;
    v8f c[4] = {};
    for (int ks = 0; ks < 8; ++ks) {
      v16h A = load_frag(w1 + (size_t)mt * 16 * 256 + ks * 32, 256, lane);
#pragma unroll
      for (int nt = 0; nt < 4; ++nt) {
        v16h Bf = load_frag(&s1[nt * 16][ks * 32], SR, lane);
        c[nt] = wmma_f16(A, Bf, c[nt]);
      }
    }
    for (int nt = 0; nt < 4; ++nt)
#pragma unroll
      for (int r = 0; r < 8; ++r)
        s2[nt * 16 + n_][mt * 16 + r + 8 * hi] = (_Float16)fmaxf(c[nt][r], 0.f);
  }
  __syncthreads();

  // stage 2: h2 = w2 @ s2
  for (int q = 0; q < 2; ++q) {
    int mt = wave * 2 + q;
    v8f c[4] = {};
    for (int ks = 0; ks < 8; ++ks) {
      v16h A = load_frag(w2 + (size_t)mt * 16 * 256 + ks * 32, 256, lane);
#pragma unroll
      for (int nt = 0; nt < 4; ++nt) {
        v16h Bf = load_frag(&s2[nt * 16][ks * 32], SR, lane);
        c[nt] = wmma_f16(A, Bf, c[nt]);
      }
    }
    for (int nt = 0; nt < 4; ++nt)
#pragma unroll
      for (int r = 0; r < 8; ++r)
        h2[nt * 16 + n_][mt * 16 + r + 8 * hi] = c[nt][r];
  }
  __syncthreads();

  // log_softmax over 256 channels, one t per thread (first 64 threads)
  if (tid < 64) {
    int t = tid, tg = t0 + t;
    float mx = -1e30f;
    for (int c = 0; c < 256; ++c) mx = fmaxf(mx, h2[t][c]);
    float s = 0.f;
    for (int c = 0; c < 256; ++c) s += __expf(h2[t][c] - mx);
    float lse = mx + __logf(s);
    if (tg < kT)
      for (int c = 0; c < 256; ++c)
        out[((size_t)b * 256 + c) * kT + tg] = h2[t][c] - lse;
  }
}

// ---------------------------------------------------------------------------
// host launch
// ---------------------------------------------------------------------------
static inline size_t al256(size_t x) { return (x + 255) & ~(size_t)255; }
static inline int blk(int n) { return (n + 255) / 256; }

extern "C" void kernel_launch(void* const* d_in, const int* in_sizes, int n_in,
                              void* d_out, int out_size, void* d_ws, size_t ws_size,
                              hipStream_t stream) {
  const float* lf      = (const float*)d_in[0];
  const int*   gold    = (const int*)  d_in[1];
  const float* emb     = (const float*)d_in[2];
  const float* up_w    = (const float*)d_in[3];
  const float* up_b    = (const float*)d_in[4];
  const float* proj_w  = (const float*)d_in[5];
  const float* proj_b  = (const float*)d_in[6];
  const float* dil_w   = (const float*)d_in[7];
  const float* dil_b   = (const float*)d_in[8];
  const float* skip_w  = (const float*)d_in[9];
  const float* skip_b  = (const float*)d_in[10];
  const float* res_w   = (const float*)d_in[11];
  const float* res_b   = (const float*)d_in[12];
  const float* out_w1  = (const float*)d_in[13];
  const float* out_w2  = (const float*)d_in[14];
  float* out = (float*)d_out;

  char* ws = (char*)d_ws;
  size_t off = 0;
  auto carve = [&](size_t bytes) { char* p = ws + off; off = al256(off + bytes); return p; };
  _Float16* xA    = (_Float16*)carve((size_t)kB * kTpad * 64 * 2);
  _Float16* xB    = (_Float16*)carve((size_t)kB * kTpad * 64 * 2);
  float*    cum   = (float*)   carve((size_t)kB * 256 * kTpad * 4);
  float*    cond  = (float*)   carve((size_t)kB * 3072 * kC4 * 4);
  float*    up    = (float*)   carve((size_t)kB * 80 * 400 * 4);
  _Float16* wdil  = (_Float16*)carve((size_t)kLayers * 2 * 128 * 64 * 2);
  _Float16* wskip = (_Float16*)carve((size_t)kLayers * 256 * 64 * 2);
  _Float16* wres  = (_Float16*)carve((size_t)(kLayers - 1) * 64 * 64 * 2);
  _Float16* w1h   = (_Float16*)carve((size_t)256 * 256 * 2);
  _Float16* w2h   = (_Float16*)carve((size_t)256 * 256 * 2);
  float*    sumsb = (float*)   carve(256 * 4);
  (void)ws_size; (void)in_sizes; (void)n_in; (void)out_size;

  // weight prep (f16 conversion / relayout)
  k_cvt_dil<<<blk(kLayers * 128 * 64 * 2), 256, 0, stream>>>(dil_w, wdil, kLayers * 128 * 64 * 2);
  k_cvt<<<blk(kLayers * 256 * 64), 256, 0, stream>>>(skip_w, wskip, kLayers * 256 * 64);
  k_cvt<<<blk((kLayers - 1) * 64 * 64), 256, 0, stream>>>(res_w, wres, (kLayers - 1) * 64 * 64);
  k_cvt<<<blk(256 * 256), 256, 0, stream>>>(out_w1, w1h, 256 * 256);
  k_cvt<<<blk(256 * 256), 256, 0, stream>>>(out_w2, w2h, 256 * 256);
  k_sumsb<<<1, 256, 0, stream>>>(skip_b, sumsb);

  // conditioning at 400-frame resolution
  k_up<<<blk(kB * 80 * 400), 256, 0, stream>>>(lf, up_w, up_b, up);
  k_cond<<<blk(kB * 3072 * kC4), 256, 0, stream>>>(proj_w, proj_b, up, cond);

  // activations init
  k_embed<<<blk(kB * kTpad * 64), 256, 0, stream>>>(emb, gold, xA);
  k_init_skip<<<blk(kB * 256 * kTpad), 256, 0, stream>>>(sumsb, cum);

  // 24 fused WMMA layers, ping-pong x buffers
  _Float16* xi = xA;
  _Float16* xo = xB;
  for (int i = 0; i < kLayers; ++i) {
    int dil = 1 << (i % 6);
    int has_res = (i < kLayers - 1) ? 1 : 0;
    const _Float16* wr = wres + (size_t)(has_res ? i : 0) * 64 * 64;
    const float*    rb = res_b + (size_t)(has_res ? i : 0) * 64;
    k_layer<<<dim3(kTpad / 128, kB), 256, 0, stream>>>(
        xi, xo, cond, cum,
        wdil + (size_t)i * 2 * 128 * 64, dil_b + (size_t)i * 128,
        wskip + (size_t)i * 256 * 64, wr, rb, dil, has_res, i);
    _Float16* t = xi; xi = xo; xo = t;
  }

  // output head + log_softmax
  k_out<<<dim3(kTpad / 64, kB), 256, 0, stream>>>(cum, w1h, w2h, out);
}
